// SoftArgsortTopK_26774826123367
// MI455X (gfx1250) — compile-verified, matrix operand order checked by reading, not run on hardware
//
#include <hip/hip_runtime.h>

// ---------------------------------------------------------------------------
// SoftArgsortTopK for MI455X (gfx1250, wave32, WMMA)
//   Kernel 1: per-row LDS bitonic sort + isotonic-L2 PAV (fast path: already
//             monotone since w-step = 1/reg = 100 >> typical sorted gaps)
//   Kernel 2: soft mask + class expectation via v_wmma_f32_16x16x32_bf16
//             A = exp(-|s - yk|/T) tile (16 k' x 32 r, bf16)
//             B = [rc | 1 | 0 ...]  (32 r x 16 cols, bf16)
//             C += A*B accumulates numerator (col 0) and denominator (col 1)
//             4-way r-split across waves for occupancy; LDS reduce at end.
// ---------------------------------------------------------------------------

typedef __attribute__((ext_vector_type(16))) __bf16         v16bf;
typedef __attribute__((ext_vector_type(8)))  float          v8f;
typedef __attribute__((ext_vector_type(4)))  float          f32x4;
typedef __attribute__((ext_vector_type(4)))  int            i32x4;
typedef __attribute__((ext_vector_type(4)))  unsigned short us4;
typedef __attribute__((ext_vector_type(8)))  unsigned short us8;

#define INV_REG 100.0f                       /* 1 / REG_STRENGTH          */
#define NEG_L2E_OVER_T (-14.4269504088896f)  /* -log2(e) / TEMPERATURE    */
#define RSPLIT 4                             /* r-range splits per b row  */

static __device__ __forceinline__ float fast_exp2(float x) {
#if __has_builtin(__builtin_amdgcn_exp2f)
    return __builtin_amdgcn_exp2f(x);   // v_exp_f32
#else
    return exp2f(x);
#endif
}

// Native f32 -> bf16 conversion: lowers to v_cvt_*bf16* on gfx1250 (RNE).
static __device__ __forceinline__ __bf16 to_bf16(float f) {
    return (__bf16)f;
}

// ---------------------------------------------------------------------------
// Kernel 1: one workgroup per row b. LDS: R floats (values) + R u16 (PAV
// counts) + flag.  48KB for R=8192.
// ---------------------------------------------------------------------------
__global__ void sort_pav_kernel(const float* __restrict__ scores,
                                float* __restrict__ yk_out,
                                int R, int k) {
    extern __shared__ char smem[];
    float*          vals = (float*)smem;
    unsigned short* cnt  = (unsigned short*)(smem + (size_t)R * 4);
    int*            flag = (int*)(smem + (size_t)R * 6);

    const int b   = blockIdx.x;
    const int tid = threadIdx.x;
    const int nth = blockDim.x;

    for (int i = tid * 4; i < R; i += nth * 4)
        *(f32x4*)(vals + i) = *(const f32x4*)(scores + (size_t)b * R + i);
    if (tid == 0) *flag = 0;
    __syncthreads();

    // Bitonic sort ascending (R is a power of two).
    for (int kk = 2; kk <= R; kk <<= 1) {
        for (int j = kk >> 1; j > 0; j >>= 1) {
            for (int i = tid; i < R; i += nth) {
                const int ixj = i ^ j;
                if (ixj > i) {
                    const float a = vals[i], c = vals[ixj];
                    const bool up = ((i & kk) == 0);
                    if ((a > c) == up) { vals[i] = c; vals[ixj] = a; }
                }
            }
            __syncthreads();
        }
    }

    // Isotonic-violation check: y[i] = w[i] + vals[i] increases iff the
    // sorted gap exceeds the w step (= INV_REG).
    for (int i = tid + 1; i < R; i += nth)
        if (vals[i] - vals[i - 1] > INV_REG) *flag = 1;
    __syncthreads();

    if (*flag == 0) {
        // Fast path: PAV solution == input, sol - w == sorted values.
        for (int j = tid; j < k; j += nth)
            yk_out[(size_t)b * k + j] = vals[R - k + j];
        return;
    }

    if (tid != 0) return;
    // Sequential pool-adjacent-violators (nonincreasing). Block means
    // overwrite vals[] in place (block index t <= scan index i).
    int t = 0;
    for (int i = 0; i < R; ++i) {
        const float y = (float)(R - i) * INV_REG + vals[i];
        vals[t] = y;
        cnt[t]  = 1;
        while (t > 0 && vals[t - 1] < vals[t]) {
            const float tot = vals[t - 1] * (float)cnt[t - 1] +
                              vals[t]     * (float)cnt[t];
            const int cc = (int)cnt[t - 1] + (int)cnt[t];
            vals[t - 1] = tot / (float)cc;
            cnt[t - 1]  = (unsigned short)cc;
            --t;
        }
        ++t;
    }
    // Expand the last k entries: yk[j] = sol[R-k+j] - w[R-k+j].
    int idx = R;
    for (int bb = t - 1; bb >= 0 && idx > R - k; --bb) {
        const int start = idx - (int)cnt[bb];
        const int lo = (start > R - k) ? start : (R - k);
        for (int i = idx - 1; i >= lo; --i)
            yk_out[(size_t)b * k + (i - (R - k))] =
                vals[bb] - (float)(R - i) * INV_REG;
        idx = start;
    }
}

// ---------------------------------------------------------------------------
// Kernel 2: one workgroup per row b; wave w -> (m-chunk, r-split).
// All exponents are <= 0 (max ~ 0 since yk lies in the score range), so the
// softmax needs no max-subtraction; result = num/den from one WMMA chain.
// ---------------------------------------------------------------------------
__global__ void softmask_wmma_kernel(const float* __restrict__ scores,
                                     const int* __restrict__ ref_class,
                                     const float* __restrict__ yk,
                                     float* __restrict__ out_sle,
                                     int R, int k, int kp, int nwm) {
    extern __shared__ char smem[];
    float*          s_row = (float*)smem;                            // R f32
    unsigned short* s_rc  = (unsigned short*)(smem + (size_t)R * 4); // R bf16
    float*          s_yk  = (float*)(smem + (size_t)R * 6);         // kp
    float*          s_num = s_yk + kp;                               // RSPLIT*kp
    float*          s_den = s_num + RSPLIT * kp;                     // RSPLIT*kp

    const int b   = blockIdx.x;
    const int tid = threadIdx.x;
    const int nth = blockDim.x;

    for (int i = tid * 4; i < R; i += nth * 4) {
        *(f32x4*)(s_row + i) = *(const f32x4*)(scores + (size_t)b * R + i);
        const i32x4 rcv = *(const i32x4*)(ref_class + i);
        us4 rb;
#pragma unroll
        for (int q = 0; q < 4; ++q)
            rb[q] = __builtin_bit_cast(unsigned short, to_bf16((float)rcv[q]));
        *(us4*)(s_rc + i) = rb;
    }
    for (int i = tid; i < kp; i += nth)
        s_yk[i] = (i < k) ? yk[(size_t)b * k + i] : 0.0f;
    __syncthreads();

    const int wave  = tid >> 5;
    const int lane  = tid & 31;
    const int mch   = wave % nwm;       // which block of 16 k'-rows
    const int split = wave / nwm;       // which quarter of the r range
    const int m0    = mch * 16;
    const int M     = lane & 15;        // A: row index; B: column index
    const int h     = lane >> 4;        // K-half selector (16-bit A layout)
    const int ko    = h * 8;            // lane's base K inside each 16-half
    const float ym  = s_yk[m0 + M];

    // Branch-free B-column selectors: col 0 = rc, col 1 = 1.0, rest 0.
    const unsigned short rc_keep  = (M == 0) ? (unsigned short)0xFFFFu : 0;
    const unsigned short one_bits = (M == 1) ? (unsigned short)0x3F80u : 0;

    const int chunks = R >> 5;                       // 32-wide r chunks
    const int c0 = (chunks * split)       / RSPLIT;
    const int c1 = (chunks * (split + 1)) / RSPLIT;

    v8f acc = {};
    for (int c = c0; c < c1; ++c) {
        const int r0 = c << 5;
        // Each lane touches two contiguous 8-element runs: wide LDS loads.
        const float* rowp = s_row + r0 + ko;
        const f32x4 s0 = *(const f32x4*)(rowp);
        const f32x4 s1 = *(const f32x4*)(rowp + 4);
        const f32x4 s2 = *(const f32x4*)(rowp + 16);
        const f32x4 s3 = *(const f32x4*)(rowp + 20);
        const us8 rc0 = *(const us8*)(s_rc + r0 + ko);
        const us8 rc1 = *(const us8*)(s_rc + r0 + 16 + ko);

        v16bf a, bm;
#pragma unroll
        for (int j = 0; j < 8; ++j) {
            // v16bf element j (vgpr j/2, half j&1) <-> K = ko + j (run 0)
            // element j+8                           <-> K = 16 + ko + j
            const float sA = (j < 4) ? s0[j & 3] : s1[j & 3];
            const float sB = (j < 4) ? s2[j & 3] : s3[j & 3];
            const float eA = fast_exp2(NEG_L2E_OVER_T * __builtin_fabsf(sA - ym));
            const float eB = fast_exp2(NEG_L2E_OVER_T * __builtin_fabsf(sB - ym));
            a[j]     = to_bf16(eA);
            a[j + 8] = to_bf16(eB);
            bm[j]     = __builtin_bit_cast(
                __bf16, (unsigned short)((rc0[j] & rc_keep) | one_bits));
            bm[j + 8] = __builtin_bit_cast(
                __bf16, (unsigned short)((rc1[j] & rc_keep) | one_bits));
        }
        acc = __builtin_amdgcn_wmma_f32_16x16x32_bf16(
            /*neg_a=*/false, a, /*neg_b=*/false, bm,
            /*c_mod=*/(short)0, acc, /*reuse_a=*/false, /*reuse_b=*/false);
    }

    // 32-bit 16x16 C/D layout: VGPR p -> row M = p + 8*h, col N = lane%16.
    if (M < 2) {
        float* dst = (M == 0) ? s_num : s_den;
        const int base = split * kp + m0 + h * 8;
#pragma unroll
        for (int p = 0; p < 8; ++p) dst[base + p] = acc[p];
    }
    __syncthreads();

    for (int i = tid; i < k; i += nth) {
        float num = 0.0f, den = 0.0f;
#pragma unroll
        for (int s = 0; s < RSPLIT; ++s) {
            num += s_num[s * kp + i];
            den += s_den[s * kp + i];
        }
        out_sle[(size_t)b * k + i] = num / den;
    }
}

// ---------------------------------------------------------------------------
extern "C" void kernel_launch(void* const* d_in, const int* in_sizes, int n_in,
                              void* d_out, int out_size, void* d_ws, size_t ws_size,
                              hipStream_t stream) {
    const float* scores    = (const float*)d_in[0];
    const int*   ref_class = (const int*)d_in[1];
    // d_in[2] = k, d_in[3] = K live on device; recover k from shapes instead:
    const int R = in_sizes[1];
    const int B = in_sizes[0] / R;
    const int k = out_size / (2 * B);     // out = yk[B,k] ++ sle[B,k]

    float* yk  = (float*)d_out;
    float* sle = (float*)d_out + (size_t)B * k;

    const size_t ldsA = (size_t)R * 6 + 16;   // vals + cnt + flag
    sort_pav_kernel<<<B, 256, ldsA, stream>>>(scores, yk, R, k);

    const int nwm = (k + 15) / 16;            // m-chunks of 16 k'-rows
    const int kp  = nwm * 16;
    const int threads = 32 * nwm * RSPLIT;    // k=32 -> 256 threads (8 waves)
    const size_t ldsB = (size_t)R * 6 + (size_t)kp * 4 +
                        (size_t)RSPLIT * kp * 8 + 16;
    softmask_wmma_kernel<<<B, threads, ldsB, stream>>>(
        scores, ref_class, yk, sle, R, k, kp, nwm);
}
